// FrameDenoisingLayer_87935160418332
// MI455X (gfx1250) — compile-verified, matrix operand order checked by reading, not run on hardware
//
#include <hip/hip_runtime.h>
#include <hip/hip_bf16.h>

// ---------------- problem constants ----------------
#define NN      2048
#define CHAIN   512
#define HH      128
#define CIN     131
#define CINP    132          // padded fused row stride (floats)
#define KPAD_A  320          // 294 -> 320 (10 k-tiles of 32)
#define KPAD_V  160          // 131 -> 160 (5 k-tiles of 32)
#define NHD     8
#define KNNK    30
#define LRK     10
#define E_KNN   (NN*KNNK)            // 61440
#define E_LR    (NN*LRK)             // 20480
#define E_SEQA  (E_KNN+E_LR)         // 81920
#define E_SEQB  (E_SEQA+NN)          // 83968
#define E_TOT   (E_SEQB+NN)          // 86016
#define NE      42                   // incident edges per dst node (max)
#define NE_PAD  48
#define NEGF    (-1e30f)

typedef __attribute__((ext_vector_type(16))) _Float16 v16h;
typedef __attribute__((ext_vector_type(8)))  float    v8f;
typedef __attribute__((ext_vector_type(4)))  unsigned int u32x4;
typedef __attribute__((ext_vector_type(8)))  int      i32x8;
typedef __attribute__((ext_vector_type(4)))  int      i32x4;

__device__ __constant__ float BBC[3][3] = {
    {-0.525f, 1.363f, 0.0f}, {0.0f, 0.0f, 0.0f}, {1.526f, 0.0f, 0.0f}};

// ---- WMMA fragment helpers (wave32 layouts per CDNA5 ISA 7.12.2) ----
// 16-bit A (16x32) / transposed-B (row = output column, 32 k-values):
//   lane L: row = L&15, hi = L>>4 ; half h: vg=h>>1,
//   k = (vg&4)*4 + (vg&3)*2 + (h&1) + hi*8
__device__ __forceinline__ v16h ldfrag(const _Float16* row, int kbase, int hi) {
  v16h f;
#pragma unroll
  for (int h = 0; h < 16; ++h) {
    int vg = h >> 1;
    int k = ((vg & 4) << 2) + ((vg & 3) << 1) + (h & 1) + (hi << 3);
    f[h] = row[kbase + k];
  }
  return f;
}

__device__ __forceinline__ v8f wmma_f16(v16h a, v16h b, v8f c) {
  return __builtin_amdgcn_wmma_f32_16x16x32_f16(false, a, false, b, (short)0, c,
                                                false, false);
}

__device__ __forceinline__ unsigned wang(unsigned s) {
  s = (s ^ 61u) ^ (s >> 16); s *= 9u; s ^= s >> 4;
  s *= 0x27d4eb2du; s ^= s >> 15; return s;
}

__device__ __forceinline__ int nxt_of(int j) {
  return (((j + 1) & (CHAIN - 1)) == 0) ? j : j + 1;
}

// ---------------- K0a: weight transpose/pad -> f16 ----------------
__global__ void k_prep_w(const float* __restrict__ Wa1, const float* __restrict__ Wa2,
                         const float* __restrict__ Wv, _Float16* Wa1T,
                         _Float16* Wa2T, _Float16* WvT) {
  const int total = 128 * KPAD_A + 16 * 128 + 256 * KPAD_V;
  for (int idx = blockIdx.x * blockDim.x + threadIdx.x; idx < total;
       idx += gridDim.x * blockDim.x) {
    if (idx < 128 * KPAD_A) {
      int n = idx / KPAD_A, k = idx % KPAD_A;
      Wa1T[idx] = (_Float16)((k < 294) ? Wa1[k * 128 + n] : 0.f);
    } else if (idx < 128 * KPAD_A + 16 * 128) {
      int r = idx - 128 * KPAD_A;
      int n = r / 128, k = r % 128;
      Wa2T[r] = (_Float16)((n < 8) ? Wa2[k * 8 + n] : 0.f);
    } else {
      int r = idx - 128 * KPAD_A - 16 * 128;
      int n = r / KPAD_V, k = r % KPAD_V;
      WvT[r] = (_Float16)((k < CIN) ? Wv[k * 256 + n] : 0.f);
    }
  }
}

// ---------------- K0b: fused node features (N,4,132) ----------------
__global__ void k_prep_fused(const float* __restrict__ nf, const float* __restrict__ rots,
                             const float* __restrict__ trans,
                             const unsigned char* __restrict__ xm,
                             const unsigned char* __restrict__ nm, float* fused) {
  int i = blockIdx.x;
  for (int qc = threadIdx.x; qc < 4 * CINP; qc += blockDim.x) {
    int q = qc / CINP, c = qc % CINP;
    float v = 0.f;
    if (c < HH) {
      v = nf[i * 512 + q * 128 + c];
    } else if (c < CIN) {
      if (q == 0) {
        v = (c == 130) ? (((nm[i] != 0) && (xm[i] == 0)) ? 1.f : 0.f) : 0.f;
      } else {
        int a = c - HH;  // backbone atom index
        const float* Rr = rots + i * 9 + (q - 1) * 3;
        v = Rr[0] * BBC[a][0] + Rr[1] * BBC[a][1] + Rr[2] * BBC[a][2] +
            trans[i * 3 + (q - 1)];
      }
    }
    fused[i * 4 * CINP + q * CINP + c] = v;
  }
}

// ---------------- K1: KNN(30) + gumbel-LR(10) per node ----------------
__global__ __launch_bounds__(256) void k_edges(const float* __restrict__ trans,
                                               int* __restrict__ esrc) {
  __shared__ float d2b[CHAIN];
  __shared__ float scb[CHAIN];
  __shared__ float rv[256];
  __shared__ int   ri[256];
  int i = blockIdx.x, tid = threadIdx.x;
  int base = i & ~(CHAIN - 1);
  float tx = trans[i * 3], ty = trans[i * 3 + 1], tz = trans[i * 3 + 2];
  for (int j = tid; j < CHAIN; j += 256) {
    int n = base + j;
    float dx = trans[n * 3] - tx, dy = trans[n * 3 + 1] - ty, dz = trans[n * 3 + 2] - tz;
    float d2 = dx * dx + dy * dy + dz * dz;
    if (n == i) d2 = 1e30f;
    d2b[j] = d2;
    unsigned h = wang((unsigned)(i * 2048 + n) ^ 42u);
    float u = ((h >> 8) + 0.5f) * (1.f / 16777216.f);
    float g = -logf(-logf(u));
    scb[j] = -1.5f * logf(d2 + 1e-30f) + g;
  }
  __syncthreads();
  // 30 x argmin
  for (int k = 0; k < KNNK; ++k) {
    float v0 = d2b[tid], v1 = d2b[tid + 256];
    float bv; int bi;
    if (v0 <= v1) { bv = v0; bi = tid; } else { bv = v1; bi = tid + 256; }
    rv[tid] = bv; ri[tid] = bi;
    __syncthreads();
    for (int s = 128; s > 0; s >>= 1) {
      if (tid < s && rv[tid + s] < rv[tid]) { rv[tid] = rv[tid + s]; ri[tid] = ri[tid + s]; }
      __syncthreads();
    }
    if (tid == 0) { esrc[i * 40 + k] = base + ri[0]; d2b[ri[0]] = 1e30f; }
    __syncthreads();
  }
  // 10 x argmax on perturbed log-distance score
  for (int k = 0; k < LRK; ++k) {
    float v0 = scb[tid], v1 = scb[tid + 256];
    float bv; int bi;
    if (v0 >= v1) { bv = v0; bi = tid; } else { bv = v1; bi = tid + 256; }
    rv[tid] = bv; ri[tid] = bi;
    __syncthreads();
    for (int s = 128; s > 0; s >>= 1) {
      if (tid < s && rv[tid + s] > rv[tid]) { rv[tid] = rv[tid + s]; ri[tid] = ri[tid + s]; }
      __syncthreads();
    }
    if (tid == 0) { esrc[i * 40 + KNNK + k] = base + ri[0]; scb[ri[0]] = -1e31f; }
    __syncthreads();
  }
}

// decode edge id -> (src,dst)
__device__ __forceinline__ void edge_sd(int e, const int* esrc, int& s, int& d) {
  if (e < E_KNN) { d = e / KNNK; s = esrc[d * 40 + e % KNNK]; }
  else if (e < E_SEQA) { int t = e - E_KNN; d = t / LRK; s = esrc[d * 40 + KNNK + t % LRK]; }
  else if (e < E_SEQB) { int j = e - E_SEQA; s = j; d = nxt_of(j); }
  else { int j = e - E_SEQB; d = j; s = nxt_of(j); }
}

// ---------------- K2: per-edge attention logits via WMMA ----------------
// 1 wave = 16-edge tile.  a_in(16x320 f16) @ Wa1T -> leaky_relu -> @ Wa2T
// Wa2T is staged into LDS via the Tensor Data Mover, overlapped with the
// A-tile build + first GEMM.
__global__ __launch_bounds__(128) void k_logits(const float* __restrict__ fused,
                                                const float* __restrict__ trans,
                                                const int* __restrict__ esrc,
                                                const _Float16* __restrict__ Wa1T,
                                                const _Float16* __restrict__ Wa2T,
                                                const float* __restrict__ ba1,
                                                float* __restrict__ logits_g) {
  __shared__ _Float16 Abuf[4][16 * KPAD_A];
  __shared__ _Float16 Wa2Lds[16 * 128];
  int wave = threadIdx.x >> 5, lane = threadIdx.x & 31;
  int n = lane & 15, hi = lane >> 4;

#if defined(__gfx1250__) && __has_builtin(__builtin_amdgcn_tensor_load_to_lds)
  if (threadIdx.x < 32) {   // wave 0 issues one TDM op; EXEC is ignored by TDM
    unsigned lds_lo = (unsigned)(size_t)(void*)&Wa2Lds[0];   // low 32b = LDS offset
    unsigned long long ga = (unsigned long long)(size_t)Wa2T;
    // D# group0: count=1, lds_addr, global_addr, type=2
    u32x4 g0 = {1u, lds_lo, (unsigned)(ga & 0xffffffffu),
                (unsigned)((ga >> 32) & 0x01ffffffu) | (2u << 30)};
    // D# group1: 1-D copy, data_size=1B, tensor_dim0=tile_dim0=4096 bytes
    i32x8 g1 = {0,
                (int)(4096u << 16),          // tensor_dim0[15:0] -> bits 63:48
                (int)(0u | (1u << 16)),      // tensor_dim0[31:16]=0, tensor_dim1=1
                (int)(4096u << 16),          // tile_dim0 -> bits 127:112
                0,
                4096,                        // tensor_dim0_stride[31:0]
                0, 0};
    i32x4 gz = {0, 0, 0, 0};
#if __clang_major__ >= 23
    i32x8 gz8 = {0, 0, 0, 0, 0, 0, 0, 0};
    __builtin_amdgcn_tensor_load_to_lds(g0, g1, gz, gz, gz8, 0);
#else
    __builtin_amdgcn_tensor_load_to_lds(g0, g1, gz, gz, 0);
#endif
  }
#else
  for (int t = threadIdx.x; t < 16 * 128; t += 128) Wa2Lds[t] = Wa2T[t];
#endif

  int ebase = (blockIdx.x * 4 + wave) * 16;
  int e = ebase + n;  // row n's edge
  int src, dst;
  edge_sd(e, esrc, src, dst);
  float dx = trans[src * 3] - trans[dst * 3];
  float dy = trans[src * 3 + 1] - trans[dst * 3 + 1];
  float dz = trans[src * 3 + 2] - trans[dst * 3 + 2];
  float dist = sqrtf(dx * dx + dy * dy + dz * dz + 1e-12f);
  float rel = (float)(src - dst);
  const float* fs = fused + src * 4 * CINP;  // row 0 = x0
  const float* fd = fused + dst * 4 * CINP;
  _Float16* Arow = &Abuf[wave][n * KPAD_A];
  for (int cc = 0; cc < 160; ++cc) {
    int c = hi * 160 + cc;
    float v;
    if (c < CIN)            v = fs[c];
    else if (c < 2 * CIN)   v = fd[c - CIN];
    else if (c < 2 * CIN + 16) {  // RBF: mu=linspace(0,20,16), sigma=1.25
      int k = c - 2 * CIN;
      float t = (dist - (float)k * (20.f / 15.f)) * 0.8f;
      v = __expf(-t * t);
    } else if (c < 2 * CIN + 32) {  // relpos
      int k = c - (2 * CIN + 16);
      int kk = k & 7;
      float ang = rel * __expf(-1.1512925465f * (float)kk);
      v = (k < 8) ? __cosf(ang) : __sinf(ang);
    } else v = 0.f;
    Arow[c] = (_Float16)v;
  }
  __syncthreads();

  v8f acc[8];
#pragma unroll
  for (int nt = 0; nt < 8; ++nt) {
    float b = ba1[nt * 16 + n];
#pragma unroll
    for (int j = 0; j < 8; ++j) acc[nt][j] = b;
  }
  for (int kt = 0; kt < 10; ++kt) {
    v16h a = ldfrag(&Abuf[wave][n * KPAD_A], kt * 32, hi);
#pragma unroll
    for (int nt = 0; nt < 8; ++nt) {
      v16h b = ldfrag(Wa1T + (nt * 16 + n) * KPAD_A, kt * 32, hi);
      acc[nt] = wmma_f16(a, b, acc[nt]);
    }
  }
  // leaky relu, stash as f16 for the second GEMM.
  // Hbuf aliases the front of this wave's Abuf region: all Abuf reads are
  // complete (same-wave LDS ordering), so reuse it and save 16KB of LDS.
  _Float16* Hbuf = &Abuf[wave][0];
#pragma unroll
  for (int nt = 0; nt < 8; ++nt) {
    int col = nt * 16 + n;
#pragma unroll
    for (int r8 = 0; r8 < 8; ++r8) {
      int m = hi * 8 + r8;
      float x = acc[nt][r8];
      Hbuf[m * HH + col] = (_Float16)(x > 0.f ? x : 0.01f * x);
    }
  }
#if defined(__gfx1250__) && __has_builtin(__builtin_amdgcn_tensor_load_to_lds) && \
    __has_builtin(__builtin_amdgcn_s_wait_tensorcnt)
  if (threadIdx.x < 32) __builtin_amdgcn_s_wait_tensorcnt(0);
#endif
  __syncthreads();
  v8f acc2;
#pragma unroll
  for (int j = 0; j < 8; ++j) acc2[j] = 0.f;
  for (int kt = 0; kt < 4; ++kt) {
    v16h a = ldfrag(&Hbuf[n * HH], kt * 32, hi);
    v16h b = ldfrag(&Wa2Lds[n * 128], kt * 32, hi);
    acc2 = wmma_f16(a, b, acc2);
  }
  if (n < NHD) {
#pragma unroll
    for (int r8 = 0; r8 < 8; ++r8) {
      int m = hi * 8 + r8;
      logits_g[(ebase + m) * NHD + n] = acc2[r8];
    }
  }
}

// ---------------- K3: per-node softmax + WMMA messages + MLP tail ----------------
__global__ __launch_bounds__(128) void k_node(
    const float* __restrict__ fused, const float* __restrict__ trans,
    const float* __restrict__ rots, const int* __restrict__ esrc,
    const float* __restrict__ logits_g, const _Float16* __restrict__ WvT,
    const float* __restrict__ Wo, const float* __restrict__ Wg,
    const float* __restrict__ Wf1, const float* __restrict__ Wf2,
    const float* __restrict__ Wr1, const float* __restrict__ br1,
    const float* __restrict__ Wr2, const float* __restrict__ br2,
    const float* __restrict__ Wr3, const float* __restrict__ br3,
    const float* __restrict__ Wt, const unsigned char* __restrict__ nmask,
    float* __restrict__ o_rots, float* __restrict__ o_trans, float* __restrict__ o_bb) {
  __shared__ float R_s[NE_PAD * 9];
  __shared__ float attn_s[NE_PAD * 8];
  __shared__ float logit_s[NE_PAD * 8];
  __shared__ int   src_s[NE_PAD];
  __shared__ _Float16 Abuf[4][16 * KPAD_V];
  __shared__ float agg[4 * 256];
  __shared__ float ybuf[4 * 128];
  __shared__ float tmp1[256];
  __shared__ float tmp2[128];
  __shared__ float inv_s[128];
  __shared__ float gate_s[128];

  int i = blockIdx.x, tid = threadIdx.x;
  for (int t = tid; t < NE_PAD * 8; t += 128) { attn_s[t] = 0.f; }
  for (int t = tid; t < NE_PAD * 9; t += 128) R_s[t] = 0.f;
  for (int t = tid; t < 1024; t += 128) agg[t] = 0.f;
  if (tid < NE_PAD) src_s[tid] = i;
  if (tid == 0) __builtin_prefetch(Wo, 0, 1);
  __syncthreads();

  // ----- phase 1: incident edge geometry + logits -----
  if (tid < NE) {
    int s = tid, e, src; bool exists = true;
    if (s < KNNK)      { e = i * KNNK + s;                src = esrc[i * 40 + s]; }
    else if (s < 40)   { e = E_KNN + i * LRK + (s - KNNK); src = esrc[i * 40 + s]; }
    else if (s == 40)  { exists = ((i & (CHAIN - 1)) != 0); e = E_SEQA + i - 1; src = exists ? (i - 1) : i; }
    else               { e = E_SEQB + i; src = nxt_of(i); }
    src_s[s] = src;
    float dx = trans[src * 3] - trans[i * 3];
    float dy = trans[src * 3 + 1] - trans[i * 3 + 1];
    float dz = trans[src * 3 + 2] - trans[i * 3 + 2];
    float dist = sqrtf(dx * dx + dy * dy + dz * dz + 1e-12f);
    bool valid = exists && (dist > 1e-3f) && (dist < 1e20f);
    float ix = 1.f / dist;
    float nx = dx * ix, ny = dy * ix, nz = dz * ix;
    float rx, ry, rz;
    if (fabsf(nx) > 0.99f) { rx = 0.f; ry = 1.f; rz = 0.f; }
    else                   { rx = 1.f; ry = 0.f; rz = 0.f; }
    float cx = ny * rz - nz * ry, cy = nz * rx - nx * rz, cz = nx * ry - ny * rx;
    float cn = 1.f / sqrtf(cx * cx + cy * cy + cz * cz + 1e-12f);
    cx *= cn; cy *= cn; cz *= cn;             // z axis
    float yx = cy * nz - cz * ny, yy = cz * nx - cx * nz, yz = cx * ny - cy * nx;
    float* R = &R_s[s * 9];
    R[0] = nx; R[1] = ny; R[2] = nz;
    R[3] = yx; R[4] = yy; R[5] = yz;
    R[6] = cx; R[7] = cy; R[8] = cz;
#pragma unroll
    for (int h = 0; h < NHD; ++h)
      logit_s[s * 8 + h] = valid ? logits_g[e * NHD + h] : NEGF;
  }
  __syncthreads();
  // ----- softmax over incident edges, per head -----
  if (tid < NHD) {
    float m = NEGF;
    for (int s = 0; s < NE; ++s) m = fmaxf(m, logit_s[s * 8 + tid]);
    float mm = (m > NEGF * 0.5f) ? m : 0.f;
    float z = 0.f;
    for (int s = 0; s < NE; ++s) z += __expf(logit_s[s * 8 + tid] - mm);
    float iz = 1.f / (z + 1e-9f);
    for (int s = 0; s < NE; ++s)
      attn_s[s * 8 + tid] = __expf(logit_s[s * 8 + tid] - mm) * iz;
  }
  __syncthreads();

  // ----- phase 2: messages.  group = 4 edges x 4 coeffs = 16 WMMA rows -----
  int wave = tid >> 5, lane = tid & 31;
  int rr = lane & 15, hi = lane >> 4;
  for (int gg = 0; gg < 3; ++gg) {
    int g = wave + gg * 4;                 // group id 0..11
    int el = rr >> 2, q = rr & 3;
    int slot = g * 4 + el;                 // < NE_PAD
    {
      int s = src_s[slot];
      const float* F = fused + s * 4 * CINP;
      const float* R = &R_s[slot * 9];
      _Float16* Arow = &Abuf[wave][rr * KPAD_V];
      for (int cc = 0; cc < 80; ++cc) {
        int c = hi * 80 + cc;
        float v = 0.f;
        if (c < CIN) {
          if (q == 0) v = F[c];
          else v = R[(q - 1) * 3 + 0] * F[1 * CINP + c] +
                   R[(q - 1) * 3 + 1] * F[2 * CINP + c] +
                   R[(q - 1) * 3 + 2] * F[3 * CINP + c];
        }
        Arow[c] = (_Float16)v;
      }
    }
    __syncthreads();
    v16h afr[5];
#pragma unroll
    for (int kt = 0; kt < 5; ++kt)
      afr[kt] = ldfrag(&Abuf[wave][rr * KPAD_V], kt * 32, hi);
    for (int nt = 0; nt < 16; ++nt) {
      v8f acc;
#pragma unroll
      for (int j = 0; j < 8; ++j) acc[j] = 0.f;
#pragma unroll
      for (int kt = 0; kt < 5; ++kt) {
        v16h b = ldfrag(WvT + (nt * 16 + rr) * KPAD_V, kt * 32, hi);
        acc = wmma_f16(afr[kt], b, acc);
      }
      int col = nt * 16 + rr;
      int head = col >> 5;
      float vals[8];
#pragma unroll
      for (int r8 = 0; r8 < 8; ++r8) {
        int m = hi * 8 + r8;
        vals[r8] = acc[r8] * attn_s[(g * 4 + (m >> 2)) * 8 + head];
      }
#pragma unroll
      for (int kk = 0; kk < 2; ++kk) {       // R^T back-rotation, in-register
        const float* R = &R_s[(g * 4 + hi * 2 + kk) * 9];
        float m1 = vals[kk * 4 + 1], m2 = vals[kk * 4 + 2], m3 = vals[kk * 4 + 3];
        vals[kk * 4 + 1] = R[0] * m1 + R[3] * m2 + R[6] * m3;
        vals[kk * 4 + 2] = R[1] * m1 + R[4] * m2 + R[7] * m3;
        vals[kk * 4 + 3] = R[2] * m1 + R[5] * m2 + R[8] * m3;
      }
#pragma unroll
      for (int r8 = 0; r8 < 8; ++r8) {
        int coef = (hi * 8 + r8) & 3;
        atomicAdd(&agg[coef * 256 + col], vals[r8]);
      }
    }
  }
  __syncthreads();

  // ----- phase 3: node MLP tail (VALU; ~2% of total FLOPs) -----
  {
    int col = tid;
    float o0 = 0.f, o1 = 0.f, o2 = 0.f, o3 = 0.f;
    for (int k = 0; k < 256; ++k) {
      float w = Wo[k * 128 + col];
      o0 += agg[k] * w; o1 += agg[256 + k] * w;
      o2 += agg[512 + k] * w; o3 += agg[768 + k] * w;
    }
    ybuf[col] = o0; ybuf[128 + col] = o1; ybuf[256 + col] = o2; ybuf[384 + col] = o3;
  }
  __syncthreads();
  {
    int col = tid;
    float gacc = 0.f, tacc = 0.f;
    for (int k = 0; k < 128; ++k) {
      float y = ybuf[k];
      gacc += y * Wg[k * 128 + col];
      tacc += y * Wf1[k * 128 + col];
    }
    gate_s[col] = 1.f / (1.f + __expf(-gacc));
    tmp1[col] = fmaxf(tacc, 0.f);
  }
  __syncthreads();
  {
    int col = tid;
    float f = 0.f;
    for (int k = 0; k < 128; ++k) f += tmp1[k] * Wf2[k * 128 + col];
    inv_s[col] = ybuf[col] + f;           // bb row 0 == inv
  }
  __syncthreads();
  for (int j = tid; j < 256; j += 128) {
    float a = br1[j];
    for (int k = 0; k < 128; ++k) a += inv_s[k] * Wr1[k * 256 + j];
    tmp1[j] = fmaxf(a, 0.f);              // h1
  }
  __syncthreads();
  {
    int col = tid;
    float a = br2[col];
    for (int k = 0; k < 256; ++k) a += tmp1[k] * Wr2[k * 128 + col];
    tmp2[col] = fmaxf(a, 0.f);            // h2
  }
  __syncthreads();
  {
    int col = tid;
    float* bb = o_bb + i * 512;
    float gcol = gate_s[col];
    bb[col]       = inv_s[col];
    bb[128 + col] = ybuf[128 + col] * gcol;
    bb[256 + col] = ybuf[256 + col] * gcol;
    bb[384 + col] = ybuf[384 + col] * gcol;
  }
  __syncthreads();
  if (tid == 0) {
    float qv[3];
#pragma unroll
    for (int q = 0; q < 3; ++q) {
      float a = br3[q];
      for (int k = 0; k < 128; ++k) a += tmp2[k] * Wr3[k * 6 + q];
      qv[q] = a;
    }
    float w = 1.f, x = qv[0], y = qv[1], z = qv[2];
    float in = 1.f / sqrtf(w * w + x * x + y * y + z * z);
    w *= in; x *= in; y *= in; z *= in;
    float RU[9];
    RU[0] = 1.f - 2.f * (y * y + z * z); RU[1] = 2.f * (x * y - w * z); RU[2] = 2.f * (x * z + w * y);
    RU[3] = 2.f * (x * y + w * z); RU[4] = 1.f - 2.f * (x * x + z * z); RU[5] = 2.f * (y * z - w * x);
    RU[6] = 2.f * (x * z - w * y); RU[7] = 2.f * (y * z + w * x); RU[8] = 1.f - 2.f * (x * x + y * y);
    bool nmv = (nmask[i] != 0);
    float tu[3];
#pragma unroll
    for (int q = 0; q < 3; ++q) {
      float a = 0.f;
      for (int k = 0; k < 128; ++k) a += (ybuf[(1 + q) * 128 + k] * gate_s[k]) * Wt[k];
      tu[q] = a;
    }
    const float* Ro = rots + i * 9;
#pragma unroll
    for (int r = 0; r < 3; ++r)
#pragma unroll
      for (int c = 0; c < 3; ++c) {
        float v = nmv ? (Ro[r * 3 + 0] * RU[0 * 3 + c] + Ro[r * 3 + 1] * RU[1 * 3 + c] +
                         Ro[r * 3 + 2] * RU[2 * 3 + c])
                      : Ro[r * 3 + c];
        o_rots[i * 9 + r * 3 + c] = v;
      }
#pragma unroll
    for (int q = 0; q < 3; ++q)
      o_trans[i * 3 + q] = trans[i * 3 + q] + (nmv ? tu[q] : 0.f);
  }
}

// ---------------- host launcher ----------------
extern "C" void kernel_launch(void* const* d_in, const int* in_sizes, int n_in,
                              void* d_out, int out_size, void* d_ws, size_t ws_size,
                              hipStream_t stream) {
  (void)in_sizes; (void)n_in; (void)out_size; (void)ws_size;
  const float* rots  = (const float*)d_in[0];
  const float* trans = (const float*)d_in[1];
  const float* nf    = (const float*)d_in[2];
  const float* Wa1 = (const float*)d_in[3];
  const float* ba1 = (const float*)d_in[4];
  const float* Wa2 = (const float*)d_in[5];
  const float* Wv  = (const float*)d_in[6];
  const float* Wo  = (const float*)d_in[7];
  const float* Wg  = (const float*)d_in[8];
  const float* Wf1 = (const float*)d_in[9];
  const float* Wf2 = (const float*)d_in[10];
  const float* Wr1 = (const float*)d_in[11];
  const float* br1 = (const float*)d_in[12];
  const float* Wr2 = (const float*)d_in[13];
  const float* br2 = (const float*)d_in[14];
  const float* Wr3 = (const float*)d_in[15];
  const float* br3 = (const float*)d_in[16];
  const float* Wt  = (const float*)d_in[17];
  const unsigned char* xm = (const unsigned char*)d_in[19];
  const unsigned char* nm = (const unsigned char*)d_in[20];

  char* ws = (char*)d_ws;
  size_t off = 0;
  auto take = [&](size_t bytes) {
    void* p = ws + off;
    off = (off + bytes + 255) & ~(size_t)255;
    return p;
  };
  _Float16* Wa1T  = (_Float16*)take(128 * KPAD_A * sizeof(_Float16));
  _Float16* Wa2T  = (_Float16*)take(16 * 128 * sizeof(_Float16));
  _Float16* WvT   = (_Float16*)take(256 * KPAD_V * sizeof(_Float16));
  float*    fused = (float*)take((size_t)NN * 4 * CINP * sizeof(float));
  int*      esrc  = (int*)take((size_t)NN * 40 * sizeof(int));
  float*    logit = (float*)take((size_t)E_TOT * NHD * sizeof(float));

  float* outp   = (float*)d_out;
  float* o_rots = outp;                  // N*9
  float* o_trns = outp + (size_t)NN * 9; // N*3
  float* o_bb   = outp + (size_t)NN * 12; // N*4*128

  k_prep_w<<<64, 256, 0, stream>>>(Wa1, Wa2, Wv, Wa1T, Wa2T, WvT);
  k_prep_fused<<<NN, 128, 0, stream>>>(nf, rots, trans, xm, nm, fused);
  k_edges<<<NN, 256, 0, stream>>>(trans, esrc);
  k_logits<<<E_TOT / 64, 128, 0, stream>>>(fused, trans, esrc, Wa1T, Wa2T, ba1, logit);
  k_node<<<NN, 128, 0, stream>>>(fused, trans, rots, esrc, logit, WvT, Wo, Wg, Wf1,
                                 Wf2, Wr1, br1, Wr2, br2, Wr3, br3, Wt, nm, o_rots,
                                 o_trns, o_bb);
}
